// _Interactions_20590073217172
// MI455X (gfx1250) — matrix operand order repaired and checked by reading, not compiled, not run on hardware
//
#include <hip/hip_runtime.h>
#include <hip/hip_fp16.h>

#define NNODES 50000     // multiple of 16
#define NEDGES 800000    // multiple of 16
#define NFEAT  64
#define ZPAD   160       // 148 padded to multiple of 32
#define WPB    8         // waves per block (wave32)
#define BLK    256

typedef __attribute__((ext_vector_type(16))) _Float16 v16h;
typedef __attribute__((ext_vector_type(8)))  _Float16 v8h;
typedef __attribute__((ext_vector_type(4)))  _Float16 v4h;
typedef __attribute__((ext_vector_type(2)))  _Float16 h2;
typedef __attribute__((ext_vector_type(8)))  float    v8f;

union HFrag { v16h v; v8h h[2]; };

__device__ __forceinline__ v8f wmma16(v16h a, v16h b, v8f c) {
  // D = A(16x32 f16) * B(32x16 f16) + C(16x16 f32)
  return __builtin_amdgcn_wmma_f32_16x16x32_f16(false, a, false, b, (short)0, c,
                                                false, false);
}

// ---------------------------------------------------------------------------
// Weight pre-conversion: f32 -> f16, B-fragment-friendly [col][k] layout.
// eW layout: [layer(2)][mat(2: f,s)][n(64)][kp(160)], kp>=148 zero-padded.
// lin0T: [n(64)][k(64)] transposed.
// ---------------------------------------------------------------------------
__global__ void prep_weights(const float* __restrict__ lin0_w,
                             const float* __restrict__ linf_w,
                             const float* __restrict__ lins_w,
                             _Float16* __restrict__ lin0T,
                             _Float16* __restrict__ eW) {
  int idx = blockIdx.x * blockDim.x + threadIdx.x;
  const int total_e = 2 * 2 * NFEAT * ZPAD;
  if (idx < total_e) {
    int kp = idx % ZPAD;
    int n  = (idx / ZPAD) % NFEAT;
    int m  = (idx / (ZPAD * NFEAT)) % 2;
    int l  = idx / (ZPAD * NFEAT * 2);
    const float* W = m ? lins_w : linf_w;
    float v = (kp < 148) ? W[((size_t)l * 148 + kp) * NFEAT + n] : 0.0f;
    eW[idx] = (_Float16)v;
  }
  if (idx < NFEAT * NFEAT) {
    int k = idx % NFEAT, n = idx / NFEAT;
    lin0T[idx] = (_Float16)lin0_w[(size_t)k * NFEAT + n];
  }
}

// ---------------------------------------------------------------------------
// ea = relu(edge_attr @ short_w + short_b), stored f16 [E][20]
// ---------------------------------------------------------------------------
__global__ void ea_kernel(const float* __restrict__ edge_attr,
                          const float* __restrict__ short_w,
                          const float* __restrict__ short_b,
                          _Float16* __restrict__ eaH, int nE) {
  int e = blockIdx.x * blockDim.x + threadIdx.x;
  if (e >= nE) return;
  float a[5];
#pragma unroll
  for (int k = 0; k < 5; ++k) a[k] = edge_attr[(size_t)e * 5 + k];
#pragma unroll
  for (int j = 0; j < 10; ++j) {
    float v0 = short_b[2 * j], v1 = short_b[2 * j + 1];
#pragma unroll
    for (int k = 0; k < 5; ++k) {
      v0 += a[k] * short_w[k * 20 + 2 * j];
      v1 += a[k] * short_w[k * 20 + 2 * j + 1];
    }
    h2 p; p.x = (_Float16)fmaxf(v0, 0.0f); p.y = (_Float16)fmaxf(v1, 0.0f);
    *(h2*)(eaH + (size_t)e * 20 + 2 * j) = p;
  }
}

// ---------------------------------------------------------------------------
// out0 = relu(h @ lin0_w + b0) via WMMA, 16-row tiles per wave
// ---------------------------------------------------------------------------
__global__ void __launch_bounds__(BLK)
lin0_kernel(const float* __restrict__ h, const _Float16* __restrict__ w0T,
            const float* __restrict__ b0, float* __restrict__ out0, int ntiles) {
  __shared__ _Float16 wls[NFEAT * NFEAT];
  __shared__ _Float16 als[WPB * 16 * NFEAT];
  {
    const uint32_t* s32 = (const uint32_t*)w0T;
    uint32_t* d32 = (uint32_t*)wls;
    for (int i = threadIdx.x; i < NFEAT * NFEAT / 2; i += BLK) d32[i] = s32[i];
  }
  __syncthreads();

  const int lane = threadIdx.x & 31, wave = threadIdx.x >> 5;
  const int hi = lane >> 4, ln = lane & 15;
  _Float16* abase = als + wave * 16 * NFEAT;

  float bv[4];
#pragma unroll
  for (int t = 0; t < 4; ++t) bv[t] = b0[t * 16 + ln];

  for (int tile = blockIdx.x * WPB + wave; tile < ntiles; tile += gridDim.x * WPB) {
    // gather: 2 rows per iteration (lanes 0-15 row r*2, lanes 16-31 row r*2+1)
#pragma unroll
    for (int r = 0; r < 8; ++r) {
      int row = (tile << 4) + r * 2 + hi;
      float4 v = *(const float4*)(h + (size_t)row * NFEAT + ln * 4);
      v4h p; p.x = (_Float16)v.x; p.y = (_Float16)v.y;
      p.z = (_Float16)v.z; p.w = (_Float16)v.w;
      *(v4h*)(abase + (r * 2 + hi) * NFEAT + ln * 4) = p;
    }
    asm volatile("s_wait_dscnt 0" ::: "memory");

    v8f acc[4] = {};
    const _Float16* arow = abase + ln * NFEAT;
#pragma unroll
    for (int kk = 0; kk < 2; ++kk) {
      HFrag a;
      a.h[0] = *(const v8h*)(arow + kk * 32 + hi * 8);
      a.h[1] = *(const v8h*)(arow + kk * 32 + 16 + hi * 8);
#pragma unroll
      for (int t = 0; t < 4; ++t) {
        const _Float16* bc = wls + (t * 16 + ln) * NFEAT + kk * 32 + hi * 8;
        HFrag b;
        b.h[0] = *(const v8h*)bc;
        b.h[1] = *(const v8h*)(bc + 16);
        acc[t] = wmma16(a.v, b.v, acc[t]);
      }
    }
#pragma unroll
    for (int j = 0; j < 8; ++j) {
      int row = (tile << 4) + hi * 8 + j;
      float* orow = out0 + (size_t)row * NFEAT + ln;
#pragma unroll
      for (int t = 0; t < 4; ++t)
        orow[t * 16] = fmaxf(acc[t][j] + bv[t], 0.0f);
    }
  }
}

// ---------------------------------------------------------------------------
// Fused edge kernel: gather z=[x_i|x_j|ea] -> WMMA x2 mats -> sigmoid*softplus
// -> atomic scatter into agg. 16 edges per wave tile (tiles are always full).
// ---------------------------------------------------------------------------
__global__ void __launch_bounds__(BLK)
edge_conv_kernel(const float* __restrict__ x,
                 const _Float16* __restrict__ eaH,
                 const int* __restrict__ srcI,
                 const int* __restrict__ dstI,
                 const _Float16* __restrict__ eW,   // [2][64][160]: f then s
                 const float* __restrict__ bf,
                 const float* __restrict__ bs,
                 float* __restrict__ agg, int ntiles) {
  __shared__ _Float16 wls[2 * NFEAT * ZPAD];   // 40 KB
  __shared__ _Float16 zls[WPB * 16 * ZPAD];    // 40 KB
  {
    const uint32_t* s32 = (const uint32_t*)eW;
    uint32_t* d32 = (uint32_t*)wls;
    for (int i = threadIdx.x; i < 2 * NFEAT * ZPAD / 2; i += BLK) d32[i] = s32[i];
  }
  __syncthreads();

  const int lane = threadIdx.x & 31, wave = threadIdx.x >> 5;
  const int hi = lane >> 4, ln = lane & 15;
  _Float16* zbase = zls + wave * 16 * ZPAD;

  float bfv[4], bsv[4];
#pragma unroll
  for (int t = 0; t < 4; ++t) {
    bfv[t] = bf[t * 16 + ln];
    bsv[t] = bs[t * 16 + ln];
  }

  for (int tile = blockIdx.x * WPB + wave; tile < ntiles; tile += gridDim.x * WPB) {
    const int ebase = tile << 4;
    // ---- gather 16 edges into LDS z-tile (f16) ----
    // lanes 0-15: x_i (dst) -> cols 0..63 ; lanes 16-31: x_j (src) -> cols 64..127
#pragma unroll
    for (int r = 0; r < 16; ++r) {
      int e = ebase + r;
      _Float16* zr = zbase + r * ZPAD;
      int nidx = (hi == 0) ? dstI[e] : srcI[e];
      float4 v = *(const float4*)(x + (size_t)nidx * NFEAT + ln * 4);
      v4h p; p.x = (_Float16)v.x; p.y = (_Float16)v.y;
      p.z = (_Float16)v.z; p.w = (_Float16)v.w;
      *(v4h*)(zr + hi * 64 + ln * 4) = p;
      if (lane < 10) {                               // ea -> cols 128..147
        uint32_t w = ((const uint32_t*)eaH)[(size_t)e * 10 + lane];
        *(uint32_t*)(zr + 128 + 2 * lane) = w;
      } else if (lane < 16) {                        // pad 148..159
        *(uint32_t*)(zr + 128 + 2 * lane) = 0u;
      }
    }
    // rows this lane will scatter in the epilogue
    int4 dA = *(const int4*)(dstI + ebase + hi * 8);
    int4 dB = *(const int4*)(dstI + ebase + hi * 8 + 4);
    asm volatile("s_wait_dscnt 0" ::: "memory");

    // ---- 16x160 @ 160x64 for both gate (f) and filter (s) mats ----
    v8f accF[4] = {}; v8f accS[4] = {};
    const _Float16* arow = zbase + ln * ZPAD;
#pragma unroll
    for (int kk = 0; kk < 5; ++kk) {
      HFrag a;
      a.h[0] = *(const v8h*)(arow + kk * 32 + hi * 8);
      a.h[1] = *(const v8h*)(arow + kk * 32 + 16 + hi * 8);
#pragma unroll
      for (int t = 0; t < 4; ++t) {
        const _Float16* bc = wls + (t * 16 + ln) * ZPAD + kk * 32 + hi * 8;
        HFrag bF, bS;
        bF.h[0] = *(const v8h*)bc;
        bF.h[1] = *(const v8h*)(bc + 16);
        bS.h[0] = *(const v8h*)(bc + NFEAT * ZPAD);
        bS.h[1] = *(const v8h*)(bc + NFEAT * ZPAD + 16);
        accF[t] = wmma16(a.v, bF.v, accF[t]);
        accS[t] = wmma16(a.v, bS.v, accS[t]);
      }
    }

    // ---- epilogue: sigmoid(f)*softplus(s), scatter-add into agg ----
    int dr[8] = {dA.x, dA.y, dA.z, dA.w, dB.x, dB.y, dB.z, dB.w};
#pragma unroll
    for (int j = 0; j < 8; ++j) {
      float* aggRow = agg + (size_t)dr[j] * NFEAT + ln;
#pragma unroll
      for (int t = 0; t < 4; ++t) {
        float f = accF[t][j] + bfv[t];
        float s = accS[t][j] + bsv[t];
        float sg = 1.0f / (1.0f + __expf(-f));
        float sp = (s > 20.0f) ? s : __logf(1.0f + __expf(s));
        unsafeAtomicAdd(aggRow + t * 16, sg * sp);
      }
    }
  }
}

// ---------------------------------------------------------------------------
// Per-channel sum / sumsq of agg (for BatchNorm batch stats)
// ---------------------------------------------------------------------------
__global__ void __launch_bounds__(256)
stats_kernel(const float* __restrict__ agg, float* __restrict__ stats, int n) {
  __shared__ float red[512];
  int tid = threadIdx.x;
  float s = 0.f, s2 = 0.f;
  for (int i = blockIdx.x * 256 + tid; i < n; i += gridDim.x * 256) {
    float v = agg[i];
    s += v; s2 += v * v;
  }
  red[tid] = s; red[256 + tid] = s2;
  __syncthreads();
  if (tid < 64) {
    float as = 0.f, as2 = 0.f;
#pragma unroll
    for (int k = 0; k < 4; ++k) { as += red[tid + 64 * k]; as2 += red[256 + tid + 64 * k]; }
    unsafeAtomicAdd(&stats[tid], as);
    unsafeAtomicAdd(&stats[64 + tid], as2);
  }
}

// ---------------------------------------------------------------------------
// out_next = out + relu( BN(agg) + out )
// ---------------------------------------------------------------------------
__global__ void __launch_bounds__(256)
bn_res_kernel(const float* __restrict__ outPrev, const float* __restrict__ agg,
              const float* __restrict__ stats, const float* __restrict__ gamma,
              const float* __restrict__ beta, float* __restrict__ outNext, int n) {
  int i = blockIdx.x * blockDim.x + threadIdx.x;
  if (i >= n) return;
  int c = i & (NFEAT - 1);
  const float invN = 1.0f / (float)NNODES;
  float mean = stats[c] * invN;
  float var  = stats[64 + c] * invN - mean * mean;
  float inv  = rsqrtf(var + 1e-5f);
  float o = outPrev[i];
  float bn = (agg[i] - mean) * inv * gamma[c] + beta[c];
  outNext[i] = o + fmaxf(bn + o, 0.0f);
}

// ---------------------------------------------------------------------------
extern "C" void kernel_launch(void* const* d_in, const int* in_sizes, int n_in,
                              void* d_out, int out_size, void* d_ws, size_t ws_size,
                              hipStream_t stream) {
  (void)in_sizes; (void)n_in; (void)out_size; (void)ws_size;
  const float* h         = (const float*)d_in[0];
  const int*   eidx      = (const int*)d_in[1];
  const float* edge_attr = (const float*)d_in[3];
  const float* lin0_w    = (const float*)d_in[4];
  const float* lin0_b    = (const float*)d_in[5];
  const float* short_w   = (const float*)d_in[6];
  const float* short_b   = (const float*)d_in[7];
  const float* linf_w    = (const float*)d_in[8];
  const float* linf_b    = (const float*)d_in[9];
  const float* lins_w    = (const float*)d_in[10];
  const float* lins_b    = (const float*)d_in[11];
  const float* bn_gamma  = (const float*)d_in[12];
  const float* bn_beta   = (const float*)d_in[13];
  const int* srcI = eidx;             // edge_index[0]
  const int* dstI = eidx + NEDGES;    // edge_index[1]

  char* p = (char*)d_ws;
  float* out0 = (float*)p; p += (size_t)NNODES * NFEAT * 4;
  float* out1 = (float*)p; p += (size_t)NNODES * NFEAT * 4;
  float* agg  = (float*)p; p += (size_t)NNODES * NFEAT * 4;
  _Float16* eaH = (_Float16*)p; p += (size_t)NEDGES * 20 * 2;
  _Float16* w0T = (_Float16*)p; p += NFEAT * NFEAT * 2;
  _Float16* eW  = (_Float16*)p; p += 2 * 2 * NFEAT * ZPAD * 2;
  float* stats  = (float*)p; p += 512;

  prep_weights<<<(2 * 2 * NFEAT * ZPAD + 255) / 256, 256, 0, stream>>>(
      lin0_w, linf_w, lins_w, w0T, eW);
  lin0_kernel<<<512, BLK, 0, stream>>>(h, w0T, lin0_b, out0, NNODES / 16);
  ea_kernel<<<(NEDGES + 255) / 256, 256, 0, stream>>>(edge_attr, short_w, short_b,
                                                      eaH, NEDGES);

  const float* xin = out0;
  for (int l = 0; l < 2; ++l) {
    float* xout = (l == 0) ? out1 : (float*)d_out;
    hipMemsetAsync(agg, 0, (size_t)NNODES * NFEAT * 4, stream);
    hipMemsetAsync(stats, 0, 128 * 4, stream);
    edge_conv_kernel<<<1024, BLK, 0, stream>>>(
        xin, eaH, srcI, dstI, eW + (size_t)l * 2 * NFEAT * ZPAD,
        linf_b + l * NFEAT, lins_b + l * NFEAT, agg, NEDGES / 16);
    stats_kernel<<<512, 256, 0, stream>>>(agg, stats, NNODES * NFEAT);
    bn_res_kernel<<<(NNODES * NFEAT + 255) / 256, 256, 0, stream>>>(
        xin, agg, stats, bn_gamma + l * NFEAT, bn_beta + l * NFEAT, xout,
        NNODES * NFEAT);
    xin = xout;
  }
}
